// TriangleUpdate_87196426043570
// MI455X (gfx1250) — matrix operand order repaired and checked by reading, not compiled, not run on hardware
//
#include <hip/hip_runtime.h>
#include <math.h>

// ---------------- CDNA5 WMMA types ----------------
typedef __attribute__((ext_vector_type(16))) __bf16 v16bf;
typedef __attribute__((ext_vector_type(8)))  __bf16 v8bf;
typedef __attribute__((ext_vector_type(4)))  __bf16 v4bf;
typedef __attribute__((ext_vector_type(8)))  float  v8f;

#define D       128
#define K3      384            // 3*D reduction depth
#define TILE_M  16             // triangles per workgroup tile
#define A_PAD   8
#define A_STRIDE (K3 + A_PAD)  // bf16 elements per LDS row (392 -> breaks bank alignment)

// ---------------------------------------------------------------------------
// Kernel 0: W (384x128 f32, k-major) -> Wt (128x384 bf16, n-major).
// Makes each lane's WMMA B fragment (16 consecutive K of one column) a single
// contiguous 32-byte global load.
// ---------------------------------------------------------------------------
__global__ void wt_convert(const float* __restrict__ W, __bf16* __restrict__ Wt) {
    int i = blockIdx.x * 256 + threadIdx.x;      // over 384*128
    if (i >= K3 * D) return;
    int k = i / D, n = i % D;
    Wt[n * K3 + k] = (__bf16)W[i];
}

// ---------------------------------------------------------------------------
// Kernel 1: persistent fused gather + bf16 WMMA GEMM + exact GELU + atomic
// segment scatter. Block = 256 threads = 8 wave32; each wave owns a 16-wide
// column panel and keeps its full 384-deep B panel in registers (12 x v16bf).
// ---------------------------------------------------------------------------
__global__ void __launch_bounds__(256)
fused_gemm_gelu_scatter(const float* __restrict__ f_edge,
                        const __bf16* __restrict__ Wt,
                        const float* __restrict__ bias,
                        const int*   __restrict__ tri,
                        float* __restrict__ segsum,   // = d_out, E*128, zeroed
                        float* __restrict__ cnt,      // E floats, zeroed
                        int T, int numTiles)
{
    __shared__ __bf16 Ash[TILE_M * A_STRIDE];
    __shared__ int    dsts[TILE_M];

    const int tid  = threadIdx.x;
    const int lane = tid & 31;
    const int wave = tid >> 5;        // 0..7 -> N sub-tile
    const int nSub = lane & 15;       // column within 16-wide tile
    const int g    = lane >> 4;       // half-wave selector (ISA A/B layouts)
    const int nCol = wave * 16 + nSub;

    // B panel resident in VGPRs for the whole kernel.
    // 32x16 bf16 B layout used here (contiguous K, not ISA table order): lane holds 16 consecutive K (g*16 .. g*16+15)
    // of its column, packed 2 per VGPR in K order -> contiguous in Wt.
    v16bf Breg[12];
#pragma unroll
    for (int kt = 0; kt < 12; ++kt)
        Breg[kt] = *(const v16bf*)(Wt + (size_t)nCol * K3 + kt * 32 + g * 16);

    const float bn = bias[nCol];

    for (int tile = blockIdx.x; tile < numTiles; tile += gridDim.x) {
        const int t0 = tile * TILE_M;

        // ---- gather: 16 rows x 384 f32 -> bf16 in LDS; record dst + count ----
        if (tid < TILE_M) {
            int t = t0 + tid;
            if (t < T) {
                int d = tri[t * 3 + 2];
                dsts[tid] = d;
                atomicAdd(&cnt[d], 1.0f);
            } else {
                dsts[tid] = -1;
            }
        }
#pragma unroll
        for (int i = tid; i < TILE_M * 96; i += 256) {   // 96 float4 per row
            int row = i / 96;
            int q   = i % 96;
            int j   = q >> 5;        // which of the 3 triangle edges
            int c4  = q & 31;        // float4 index within the 128-dim feature
            int t   = t0 + row;
            float4 v = make_float4(0.f, 0.f, 0.f, 0.f);
            if (t < T) {
                int e = tri[t * 3 + j];
                v = ((const float4*)f_edge)[e * 32 + c4];
            }
            v4bf b4 = { (__bf16)v.x, (__bf16)v.y, (__bf16)v.z, (__bf16)v.w };
            *(v4bf*)(Ash + row * A_STRIDE + q * 4) = b4;
        }
        __syncthreads();

        // ---- 16x16 tile: 12 x v_wmma_f32_16x16x32_bf16 over K=384 ----
        v8f acc = {0.f, 0.f, 0.f, 0.f, 0.f, 0.f, 0.f, 0.f};
        const int m = lane & 15;
#pragma unroll
        for (int kt = 0; kt < 12; ++kt) {
            // A layout (matches B): lane holds K = kt*32 + g*16 + [0..15]
            // in VGPR0-7, contiguous K order.
            const __bf16* base = Ash + m * A_STRIDE + kt * 32 + g * 16;
            v8bf lo = *(const v8bf*)(base);
            v8bf hi = *(const v8bf*)(base + 8);
            v16bf a;
#pragma unroll
            for (int e2 = 0; e2 < 8; ++e2) { a[e2] = lo[e2]; a[8 + e2] = hi[e2]; }
            acc = __builtin_amdgcn_wmma_f32_16x16x32_bf16(
                    false, a, false, Breg[kt], (short)0, acc, false, false);
        }

        // ---- epilogue: bias + exact GELU + f32 atomic scatter-add ----
        // C layout: VGPR r holds row (r + 8*g), col nSub.
#pragma unroll
        for (int r = 0; r < 8; ++r) {
            int mrow = r + 8 * g;
            int d = dsts[mrow];
            if (d >= 0) {
                float x  = acc[r] + bn;
                float ge = 0.5f * x * (1.0f + erff(x * 0.70710678118654752f));
                atomicAdd(&segsum[(size_t)d * D + nCol], ge);
            }
        }
        __syncthreads();   // protect LDS before the next tile's gather
    }
}

// ---------------------------------------------------------------------------
// Kernel 2: per-edge mean (sum / max(cnt,1)) + LayerNorm, in place over d_out.
// One wave per 128-dim row, float4 per lane, shfl_xor tree reductions.
// ---------------------------------------------------------------------------
__global__ void __launch_bounds__(256)
segmean_layernorm(float* __restrict__ out,
                  const float* __restrict__ cnt,
                  const float* __restrict__ gamma,
                  const float* __restrict__ beta,
                  int E)
{
    int lane = threadIdx.x & 31;
    int row  = blockIdx.x * 8 + (threadIdx.x >> 5);
    if (row >= E) return;

    float4 s = ((const float4*)out)[(size_t)row * 32 + lane];
    float inv_c = 1.0f / fmaxf(cnt[row], 1.0f);
    float m0 = s.x * inv_c, m1 = s.y * inv_c, m2 = s.z * inv_c, m3 = s.w * inv_c;

    float p = m0 + m1 + m2 + m3;
#pragma unroll
    for (int off = 16; off > 0; off >>= 1) p += __shfl_xor(p, off, 32);
    float mu = p * (1.0f / 128.0f);

    float d0 = m0 - mu, d1 = m1 - mu, d2 = m2 - mu, d3 = m3 - mu;
    float q = d0 * d0 + d1 * d1 + d2 * d2 + d3 * d3;
#pragma unroll
    for (int off = 16; off > 0; off >>= 1) q += __shfl_xor(q, off, 32);
    float inv = rsqrtf(q * (1.0f / 128.0f) + 1e-5f);

    float4 gg = ((const float4*)gamma)[lane];
    float4 bb = ((const float4*)beta)[lane];
    float4 o;
    o.x = d0 * inv * gg.x + bb.x;
    o.y = d1 * inv * gg.y + bb.y;
    o.z = d2 * inv * gg.z + bb.z;
    o.w = d3 * inv * gg.w + bb.w;
    ((float4*)out)[(size_t)row * 32 + lane] = o;
}

// ---------------------------------------------------------------------------
extern "C" void kernel_launch(void* const* d_in, const int* in_sizes, int n_in,
                              void* d_out, int out_size, void* d_ws, size_t ws_size,
                              hipStream_t stream) {
    const float* f_edge = (const float*)d_in[0];
    const float* W      = (const float*)d_in[1];
    const float* bvec   = (const float*)d_in[2];
    const float* gamma  = (const float*)d_in[3];
    const float* beta   = (const float*)d_in[4];
    const int*   tri    = (const int*)d_in[5];

    const int E = in_sizes[0] / D;
    const int T = in_sizes[5] / 3;

    float* out = (float*)d_out;
    float* cnt = (float*)d_ws;
    size_t wtOff = (((size_t)E * 4) + 255) & ~(size_t)255;
    __bf16* Wt = (__bf16*)((char*)d_ws + wtOff);

    // zero accumulators (memset nodes are graph-capturable)
    hipMemsetAsync(out, 0, (size_t)E * D * sizeof(float), stream);
    hipMemsetAsync(cnt, 0, (size_t)E * sizeof(float), stream);

    wt_convert<<<(K3 * D + 255) / 256, 256, 0, stream>>>(W, Wt);

    const int numTiles = (T + TILE_M - 1) / TILE_M;
    const int grid1 = numTiles < 2048 ? numTiles : 2048;   // persistent blocks
    fused_gemm_gelu_scatter<<<grid1, 256, 0, stream>>>(
        f_edge, Wt, bvec, tri, out, cnt, T, numTiles);

    segmean_layernorm<<<(E + 7) / 8, 256, 0, stream>>>(out, cnt, gamma, beta, E);
}
